// PointerNetwork_29300266893483
// MI455X (gfx1250) — matrix operand order repaired and checked by reading, not compiled, not run on hardware
//
#include <hip/hip_runtime.h>
#include <hip/hip_bf16.h>
#include <math.h>

// ---------------------------------------------------------------------------
// PointerNetwork forward for MI455X (gfx1250, wave32, WMMA + TDM).
//  - Encoder GEMMs: LDS-staged tiles (TDM tensor_load_to_lds when available,
//    cooperative copy fallback) feeding v_wmma_f32_16x16x32_f16.
//  - Decoder: persistent kernel, 64 blocks, LSTM/softmax state replicated in
//    LDS per block; only 2 global grid barriers per decode step.
// ---------------------------------------------------------------------------

typedef _Float16 v16h __attribute__((ext_vector_type(16)));
typedef float    v8f  __attribute__((ext_vector_type(8)));

constexpr int SEQ   = 2048;   // L
constexpr int HID   = 256;    // H
constexpr int NHEAD = 8;
constexpr int HDH   = 32;     // head dim
constexpr int DFFI  = 1024;   // 4*H
constexpr int NLAY  = 2;
constexpr int NB    = 64;     // decode persistent blocks

#if defined(__HIP_DEVICE_COMPILE__) && __has_builtin(__builtin_amdgcn_tensor_load_to_lds)
#define USE_TDM 1
#else
#define USE_TDM 0
#endif

// ------------------------------- helpers -----------------------------------

__device__ inline float sigf(float x) { return 1.0f / (1.0f + expf(-x)); }

__device__ inline float block_reduce_sum(float v, float* red) {
  int tid = threadIdx.x;
  red[tid] = v; __syncthreads();
  for (int s = 128; s > 0; s >>= 1) {
    if (tid < s) red[tid] += red[tid + s];
    __syncthreads();
  }
  float r = red[0]; __syncthreads();
  return r;
}

__device__ inline float block_reduce_max(float v, float* red) {
  int tid = threadIdx.x;
  red[tid] = v; __syncthreads();
  for (int s = 128; s > 0; s >>= 1) {
    if (tid < s) red[tid] = fmaxf(red[tid], red[tid + s]);
    __syncthreads();
  }
  float r = red[0]; __syncthreads();
  return r;
}

__device__ inline void grid_barrier(int* bar, int nb) {
  __syncthreads();
  __threadfence();
  if (threadIdx.x == 0) {
    int g = atomicAdd(&bar[1], 0);                 // read generation
    if (atomicAdd(&bar[0], 1) == nb - 1) {
      atomicExch(&bar[0], 0);
      __threadfence();
      atomicAdd(&bar[1], 1);                       // release
    } else {
      while (atomicAdd(&bar[1], 0) == g) __builtin_amdgcn_s_sleep(1);
    }
  }
  __syncthreads();
  __threadfence();
}

#if USE_TDM
typedef unsigned int u32x4 __attribute__((ext_vector_type(4)));
typedef int          i32x4 __attribute__((ext_vector_type(4)));
typedef int          i32x8 __attribute__((ext_vector_type(8)));

// 2D f16 tile DMA: global (row stride = stride_e elements) -> contiguous LDS.
// Tensor dims set equal to tile dims (tiles are always fully in-bounds).
// D# layout per CDNA5 ISA 8.3/8.4 (count=1, type=2, data_size=2B).
__device__ inline void tdm_load_2d_f16(const _Float16* gptr, unsigned lds_off,
                                       unsigned stride_e, unsigned tile_d0,
                                       unsigned tile_d1) {
  unsigned long long ga = (unsigned long long)gptr;
  u32x4 g0;
  g0[0] = 1u;                                        // count=1 (valid), user mode
  g0[1] = lds_off;                                   // lds_addr [63:32]
  g0[2] = (unsigned)(ga & 0xffffffffu);              // global_addr [95:64]
  g0[3] = (unsigned)((ga >> 32) & 0x01ffffffu)       // global_addr [120:96]
        | (2u << 30);                                // type=2 ("image")
  i32x8 g1;
  g1[0] = (int)(1u << 16);                           // data_size=1 (2 bytes)
  g1[1] = (int)((tile_d0 & 0xffffu) << 16);          // tensor_dim0[15:0]
  g1[2] = (int)((tile_d1 & 0xffffu) << 16);          // tensor_dim0[31:16]=0 | tensor_dim1[15:0]
  g1[3] = (int)((tile_d0 & 0xffffu) << 16);          // tensor_dim1[31:16]=0 | tile_dim0
  g1[4] = (int)(tile_d1 & 0xffffu);                  // tile_dim1 | tile_dim2=0
  g1[5] = (int)stride_e;                             // tensor_dim0_stride[31:0]
  g1[6] = 0;                                         // stride0[47:32]=0 | dim1_stride[15:0]=0
  g1[7] = 0;
  i32x4 gz = {0, 0, 0, 0};
#if defined(__clang_major__) && __clang_major__ >= 23
  i32x8 gz8 = {0, 0, 0, 0, 0, 0, 0, 0};
  __builtin_amdgcn_tensor_load_to_lds(g0, g1, gz, gz, gz8, 0);
#else
  __builtin_amdgcn_tensor_load_to_lds(g0, g1, gz, gz, 0);
#endif
}
#endif  // USE_TDM

// ----------------------------- small kernels -------------------------------

__global__ void cvt_f32_f16(const float* __restrict__ in, _Float16* __restrict__ out, int n) {
  int i = blockIdx.x * blockDim.x + threadIdx.x;
  if (i < n) out[i] = (_Float16)in[i];
}

__global__ void init_bar_kernel(int* bar) { bar[0] = 0; bar[1] = 0; }

__global__ void embed_kernel(const float* __restrict__ in, const float* __restrict__ W,
                             const float* __restrict__ b, float* __restrict__ x,
                             float* __restrict__ emb) {
  int l = blockIdx.x, hh = threadIdx.x;
  float v = in[l * 2 + 0] * W[hh * 2 + 0] + in[l * 2 + 1] * W[hh * 2 + 1] + b[hh];
  x[l * HID + hh] = v;
  emb[l * HID + hh] = v;
}

__global__ __launch_bounds__(256)
void ln_kernel(const float* __restrict__ x, const float* __restrict__ s,
               const float* __restrict__ b, _Float16* __restrict__ y) {
  __shared__ float red[256];
  int row = blockIdx.x, tid = threadIdx.x;
  float v = x[(size_t)row * HID + tid];
  float mean = block_reduce_sum(v, red) * (1.0f / HID);
  float d = v - mean;
  float var = block_reduce_sum(d * d, red) * (1.0f / HID);
  float r = rsqrtf(var + 1e-5f);
  y[(size_t)row * HID + tid] = (_Float16)(d * r * s[tid] + b[tid]);
}

// --------------------------- WMMA GEMM kernel ------------------------------
// C(M,N) = A(M,K) * W(N,K)^T  [+bias] [ReLU] [+resid]; out f32 and/or f16.
// Block = 8 waves = 32x64 output tile; K-step 32.
// A tile (32x32) and B tile (64x32) staged in dynamic LDS:
//   LDS[0..2048)     = tA  (offset 0 for TDM lds_addr)
//   LDS[2048..6144)  = tB
// Dynamic shared size at launch: 6144 bytes.

__global__ __launch_bounds__(256)
void gemm_wmma(const _Float16* __restrict__ A, const _Float16* __restrict__ W,
               const float* __restrict__ bias, const float* __restrict__ resid,
               float* __restrict__ Cf, _Float16* __restrict__ Ch,
               int M, int N, int K, int relu) {
  extern __shared__ _Float16 smem[];
  _Float16* tA = smem;                  // 32 rows x 32 k
  _Float16* tB = smem + 32 * 32;        // 64 rows x 32 k

  const int tid  = threadIdx.x;
  const int wave = tid >> 5;
  const int lane = tid & 31;
  const int hl   = lane >> 4;           // half-wave
  const int mblock = blockIdx.y * 32;
  const int nblock = blockIdx.x * 64;
  const int mw = (wave >> 2) * 16;      // wave row offset within block tile
  const int nw = (wave & 3) * 16;       // wave col offset within block tile

  v8f acc = {0.f, 0.f, 0.f, 0.f, 0.f, 0.f, 0.f, 0.f};

  for (int k0 = 0; k0 < K; k0 += 32) {
#if USE_TDM
    if (wave == 0) {
      tdm_load_2d_f16(A + (size_t)mblock * K + k0, 0u,    (unsigned)K, 32u, 32u);
      tdm_load_2d_f16(W + (size_t)nblock * K + k0, 2048u, (unsigned)K, 32u, 64u);
      __builtin_amdgcn_s_wait_tensorcnt(0);
    }
    __builtin_prefetch(A + (size_t)(mblock + (tid >> 3)) * K + k0 + 32, 0, 1);
#else
    for (int i = tid; i < 32 * 32; i += 256) {
      tA[i] = A[(size_t)(mblock + (i >> 5)) * K + k0 + (i & 31)];
    }
    for (int i = tid; i < 64 * 32; i += 256) {
      tB[i] = W[(size_t)(nblock + (i >> 5)) * K + k0 + (i & 31)];
    }
    __builtin_prefetch(A + (size_t)(mblock + (tid >> 3)) * K + k0 + 32, 0, 1);
    __builtin_prefetch(W + (size_t)(nblock + (tid >> 2)) * K + k0 + 32, 0, 1);
#endif
    __syncthreads();

    v16h a, bf;
    // A fragment: 16x32 f16, ISA 7.12.2 layout; B fragment: 32x16 (K-pairs).
#pragma unroll
    for (int v = 0; v < 8; ++v) {
      int kb = ((v < 4) ? 0 : 16) + hl * 8 + (v & 3) * 2;
      int ra = mw + (lane & 15);
      a[2 * v]     = tA[ra * 32 + kb];
      a[2 * v + 1] = tA[ra * 32 + kb + 1];
      int rb = nw + (lane & 15);
      int kc = ((v < 4) ? 0 : 16) + hl * 8 + (v & 3) * 2;
      bf[2 * v]     = tB[rb * 32 + kc];
      bf[2 * v + 1] = tB[rb * 32 + kc + 1];
    }
    acc = __builtin_amdgcn_wmma_f32_16x16x32_f16(
        false, a, false, bf, (short)0, acc, false, false);
    __syncthreads();
  }

#pragma unroll
  for (int j = 0; j < 8; ++j) {
    int m = mblock + mw + j + 8 * hl;
    int n = nblock + nw + (lane & 15);
    float v = acc[j];
    if (bias)  v += bias[n];
    if (relu)  v = fmaxf(v, 0.0f);
    if (resid) v += resid[(size_t)m * N + n];
    if (Cf) Cf[(size_t)m * N + n] = v;
    if (Ch) Ch[(size_t)m * N + n] = (_Float16)v;
  }
}

// ----------------------------- attention -----------------------------------
// One block (256 threads) per (query, head). qkv row = [q | k | v], 768 f32.

__global__ __launch_bounds__(256)
void attn_kernel(const float* __restrict__ qkv, const int* __restrict__ lengths,
                 _Float16* __restrict__ out) {
  __shared__ float qs[HDH];
  __shared__ float red[256];
  __shared__ float ovec[256 * HDH];           // 32 KB

  int q = blockIdx.x, h = blockIdx.y, tid = threadIdx.x;
  int len = lengths[0];
  if (tid < HDH) qs[tid] = qkv[(size_t)q * 768 + h * HDH + tid];
  __syncthreads();

  const float inv_sqrt_hd = 0.17677669529663687f;  // 1/sqrt(32)
  float s[SEQ / 256];
  float lmax = -1e30f;
#pragma unroll
  for (int i = 0; i < SEQ / 256; ++i) {
    int k = tid + i * 256;
    const float* kp = qkv + (size_t)k * 768 + HID + h * HDH;
    float dot = 0.f;
#pragma unroll
    for (int d = 0; d < HDH; ++d) dot += qs[d] * kp[d];
    dot *= inv_sqrt_hd;
    if (k >= len) dot = -1e9f;
    s[i] = dot;
    lmax = fmaxf(lmax, dot);
  }
  float m = block_reduce_max(lmax, red);
  float lsum = 0.f;
#pragma unroll
  for (int i = 0; i < SEQ / 256; ++i) { s[i] = expf(s[i] - m); lsum += s[i]; }
  float denom = block_reduce_sum(lsum, red);

  float po[HDH];
#pragma unroll
  for (int d = 0; d < HDH; ++d) po[d] = 0.f;
#pragma unroll
  for (int i = 0; i < SEQ / 256; ++i) {
    int k = tid + i * 256;
    const float* vp = qkv + (size_t)k * 768 + 2 * HID + h * HDH;
#pragma unroll
    for (int d = 0; d < HDH; ++d) po[d] += s[i] * vp[d];
  }
#pragma unroll
  for (int d = 0; d < HDH; ++d) ovec[tid * HDH + d] = po[d];
  __syncthreads();

  if (tid < HDH) {
    float o = 0.f;
    for (int j = 0; j < 256; ++j) o += ovec[j * HDH + tid];
    out[(size_t)q * HID + h * HDH + tid] = (_Float16)(o / denom);
  }
}

// ----------------------------- decode (persistent) -------------------------

__global__ __launch_bounds__(256)
void decode_kernel(const float* __restrict__ embedded, const float* __restrict__ enc,
                   const float* __restrict__ refg, const float* __restrict__ refp,
                   const float* __restrict__ Wih, const float* __restrict__ Whh,
                   const float* __restrict__ bih, const float* __restrict__ bhh,
                   const float* __restrict__ gWq, const float* __restrict__ gv,
                   const float* __restrict__ pWq, const float* __restrict__ pv,
                   const float* __restrict__ dec0, const int* __restrict__ lengths,
                   float* __restrict__ sgbuf, float* __restrict__ spbuf, int* bar,
                   int* __restrict__ tour_out, float* __restrict__ logp_out) {
  __shared__ float h_s[HID], c_s[HID], din[HID], hq[HID], ctx[HID], cq[HID];
  __shared__ float gates[4 * HID];
  __shared__ float red[256];
  __shared__ int   redi[256];
  __shared__ float pw[SEQ];                   // 8 KB: softmax probs
  __shared__ unsigned mask[SEQ / 32];
  __shared__ int s_cnt;
  __shared__ float gvs[HID], pvs[HID];

  const int tid = threadIdx.x;
  const int b   = blockIdx.x;
  const int len = lengths[0];
  const int ROWS = SEQ / NB;                  // 32 rows per block
  const float scale = 0.0625f;                // 1/sqrt(256)

  if (tid < HID) {
    h_s[tid] = 0.f; c_s[tid] = 0.f; din[tid] = dec0[tid];
    gvs[tid] = gv[tid]; pvs[tid] = pv[tid];
  }
  if (tid < SEQ / 32) {
    unsigned m = 0;
    for (int j = 0; j < 32; ++j) { int l = tid * 32 + j; if (l >= len) m |= 1u << j; }
    mask[tid] = m;
  }
  if (tid == 0) s_cnt = (SEQ > len) ? (SEQ - len) : 0;
  __syncthreads();

  for (int t = 0; t < SEQ; ++t) {
    if (tid == 0 && s_cnt >= SEQ) { mask[0] &= ~1u; s_cnt = SEQ - 1; }
    __syncthreads();

    // ---- LSTM cell (replicated per block) ----
    for (int n = tid; n < 4 * HID; n += 256) {
      const float* wi = Wih + (size_t)n * HID;
      const float* wh = Whh + (size_t)n * HID;
      float acc = bih[n] + bhh[n];
      for (int k = 0; k < HID; ++k) acc += wi[k] * din[k] + wh[k] * h_s[k];
      gates[n] = acc;
    }
    __syncthreads();
    if (tid < HID) {
      float ig = sigf(gates[tid]);
      float fg = sigf(gates[HID + tid]);
      float gg = tanhf(gates[2 * HID + tid]);
      float og = sigf(gates[3 * HID + tid]);
      float c = fg * c_s[tid] + ig * gg;
      c_s[tid] = c;
      h_s[tid] = og * tanhf(c);
    }
    __syncthreads();
    if (tid < HID) {                          // hq = h @ g_Wq^T
      const float* w = gWq + (size_t)tid * HID;
      float a = 0.f;
      for (int k = 0; k < HID; ++k) a += w[k] * h_s[k];
      hq[tid] = a;
    }
    __syncthreads();

    // ---- s_g for this block's rows ----
    {
      int rl = tid >> 3, ln = tid & 7;
      int r = b * ROWS + rl;
      const float* rg = refg + (size_t)r * HID + ln * 32;
      float p = 0.f;
#pragma unroll
      for (int d = 0; d < 32; ++d) p += tanhf(rg[d] + hq[ln * 32 + d]) * gvs[ln * 32 + d];
      for (int off = 4; off; off >>= 1) p += __shfl_down(p, off, 8);
      if (ln == 0) {
        bool mk = (mask[r >> 5] >> (r & 31)) & 1u;
        sgbuf[r] = mk ? -1e9f : p * scale;
      }
    }
    grid_barrier(bar, NB);                    // barrier #1

    // ---- redundant softmax over sgbuf + context ----
    float lv[SEQ / 256];
    float lmax = -1e30f;
#pragma unroll
    for (int i = 0; i < SEQ / 256; ++i) {
      float v = sgbuf[tid + i * 256];
      lv[i] = v; lmax = fmaxf(lmax, v);
    }
    float gm = block_reduce_max(lmax, red);
    float ls = 0.f;
#pragma unroll
    for (int i = 0; i < SEQ / 256; ++i) { lv[i] = expf(lv[i] - gm); ls += lv[i]; }
    float gs = block_reduce_sum(ls, red);
    float invs = 1.0f / gs;
#pragma unroll
    for (int i = 0; i < SEQ / 256; ++i) pw[tid + i * 256] = lv[i] * invs;
    __syncthreads();
    {
      float acc = 0.f;
      for (int l = 0; l < SEQ; ++l) acc += pw[l] * enc[(size_t)l * HID + tid];
      ctx[tid] = acc;
    }
    __syncthreads();
    if (tid < HID) {                          // cq = ctx @ p_Wq^T
      const float* w = pWq + (size_t)tid * HID;
      float a = 0.f;
      for (int k = 0; k < HID; ++k) a += w[k] * ctx[k];
      cq[tid] = a;
    }
    __syncthreads();

    // ---- s_p for this block's rows ----
    {
      int rl = tid >> 3, ln = tid & 7;
      int r = b * ROWS + rl;
      const float* rp = refp + (size_t)r * HID + ln * 32;
      float p = 0.f;
#pragma unroll
      for (int d = 0; d < 32; ++d) p += tanhf(rp[d] + cq[ln * 32 + d]) * pvs[ln * 32 + d];
      for (int off = 4; off; off >>= 1) p += __shfl_down(p, off, 8);
      if (ln == 0) {
        bool mk = (mask[r >> 5] >> (r & 31)) & 1u;
        spbuf[r] = mk ? -1e9f : p * scale;
      }
    }
    grid_barrier(bar, NB);                    // barrier #2

    // ---- redundant max/argmax/log-sum-exp over spbuf ----
    float sv[SEQ / 256];
    float bmax = -1e30f; int bidx = 0;
#pragma unroll
    for (int i = 0; i < SEQ / 256; ++i) {
      int l = tid + i * 256;
      float v = spbuf[l];
      sv[i] = v;
      if (v > bmax) { bmax = v; bidx = l; }
    }
    red[tid] = bmax; redi[tid] = bidx; __syncthreads();
    for (int s = 128; s > 0; s >>= 1) {
      if (tid < s) {
        float ov = red[tid + s]; int oi = redi[tid + s];
        if (ov > red[tid] || (ov == red[tid] && oi < redi[tid])) {
          red[tid] = ov; redi[tid] = oi;
        }
      }
      __syncthreads();
    }
    float pmax = red[0]; int pidx = redi[0];
    __syncthreads();
    float es = 0.f;
#pragma unroll
    for (int i = 0; i < SEQ / 256; ++i) es += expf(sv[i] - pmax);
    float tot = block_reduce_sum(es, red);
    float lse = pmax + logf(tot);

    if (tid < ROWS) {
      int r = b * ROWS + tid;
      logp_out[(size_t)t * SEQ + r] = spbuf[r] - lse;
    }
    if (b == 0 && tid == 0) tour_out[t] = pidx;

    if (tid == 0) { mask[pidx >> 5] |= 1u << (pidx & 31); s_cnt++; }
    if (tid < HID) din[tid] = embedded[(size_t)pidx * HID + tid];
    __syncthreads();
  }
}

// ------------------------------- launcher ----------------------------------

extern "C" void kernel_launch(void* const* d_in, const int* in_sizes, int n_in,
                              void* d_out, int out_size, void* d_ws, size_t ws_size,
                              hipStream_t stream) {
  (void)in_sizes; (void)n_in; (void)out_size; (void)ws_size;

  const float* inputs    = (const float*)d_in[0];
  const int*   lengths   = (const int*)d_in[1];
  const float* emb_W     = (const float*)d_in[2];
  const float* emb_b     = (const float*)d_in[3];
  const float* ln1_s     = (const float*)d_in[4];
  const float* ln1_b     = (const float*)d_in[5];
  const float* attn_in_W = (const float*)d_in[6];
  const float* attn_in_b = (const float*)d_in[7];
  const float* attn_out_W= (const float*)d_in[8];
  const float* attn_out_b= (const float*)d_in[9];
  const float* ln2_s     = (const float*)d_in[10];
  const float* ln2_b     = (const float*)d_in[11];
  const float* ffn_W1    = (const float*)d_in[12];
  const float* ffn_b1    = (const float*)d_in[13];
  const float* ffn_W2    = (const float*)d_in[14];
  const float* ffn_b2    = (const float*)d_in[15];
  const float* lstm_Wih  = (const float*)d_in[16];
  const float* lstm_Whh  = (const float*)d_in[17];
  const float* lstm_bih  = (const float*)d_in[18];
  const float* lstm_bhh  = (const float*)d_in[19];
  const float* g_Wref    = (const float*)d_in[20];
  const float* g_Wq      = (const float*)d_in[21];
  const float* g_v       = (const float*)d_in[22];
  const float* p_Wref    = (const float*)d_in[23];
  const float* p_Wq      = (const float*)d_in[24];
  const float* p_v       = (const float*)d_in[25];
  const float* dec0      = (const float*)d_in[26];

  char* ws = (char*)d_ws;
  size_t off = 0;
  auto alloc = [&](size_t bytes) -> void* {
    void* p = ws + off;
    off += (bytes + 255) & ~(size_t)255;
    return p;
  };

  float*    x       = (float*)alloc((size_t)SEQ * HID * 4);
  float*    emb     = (float*)alloc((size_t)SEQ * HID * 4);
  float*    qkv     = (float*)alloc((size_t)SEQ * 3 * HID * 4);
  _Float16* attn_oh = (_Float16*)alloc((size_t)SEQ * HID * 2);
  _Float16* y_h     = (_Float16*)alloc((size_t)SEQ * HID * 2);
  _Float16* hid_h   = (_Float16*)alloc((size_t)SEQ * DFFI * 2);
  _Float16* enc_h   = (_Float16*)alloc((size_t)SEQ * HID * 2);
  float*    ref_g   = (float*)alloc((size_t)SEQ * HID * 4);
  float*    ref_p   = (float*)alloc((size_t)SEQ * HID * 4);
  float*    sgbuf   = (float*)alloc((size_t)SEQ * 4);
  float*    spbuf   = (float*)alloc((size_t)SEQ * 4);
  int*      bar     = (int*)alloc(256);
  _Float16* Wqkv_h  = (_Float16*)alloc((size_t)NLAY * 3 * HID * HID * 2);
  _Float16* Wout_h  = (_Float16*)alloc((size_t)NLAY * HID * HID * 2);
  _Float16* W1_h    = (_Float16*)alloc((size_t)NLAY * DFFI * HID * 2);
  _Float16* W2_h    = (_Float16*)alloc((size_t)NLAY * HID * DFFI * 2);
  _Float16* gWref_h = (_Float16*)alloc((size_t)HID * HID * 2);
  _Float16* pWref_h = (_Float16*)alloc((size_t)HID * HID * 2);

  int*   tour = (int*)d_out;
  float* logp = (float*)d_out + SEQ;

  auto cvt = [&](const float* src, _Float16* dst, int n) {
    cvt_f32_f16<<<(n + 255) / 256, 256, 0, stream>>>(src, dst, n);
  };
  cvt(attn_in_W,  Wqkv_h, NLAY * 3 * HID * HID);
  cvt(attn_out_W, Wout_h, NLAY * HID * HID);
  cvt(ffn_W1,     W1_h,   NLAY * DFFI * HID);
  cvt(ffn_W2,     W2_h,   NLAY * HID * DFFI);
  cvt(g_Wref,     gWref_h, HID * HID);
  cvt(p_Wref,     pWref_h, HID * HID);

  embed_kernel<<<SEQ, HID, 0, stream>>>(inputs, emb_W, emb_b, x, emb);

  const size_t GEMM_LDS = (32 * 32 + 64 * 32) * sizeof(_Float16);  // 6144 B

  for (int l = 0; l < NLAY; ++l) {
    // ln1 -> y_h
    ln_kernel<<<SEQ, HID, 0, stream>>>(x, ln1_s + l * HID, ln1_b + l * HID, y_h);
    // qkv = y @ Win^T + b
    gemm_wmma<<<dim3(3 * HID / 64, SEQ / 32), 256, GEMM_LDS, stream>>>(
        y_h, Wqkv_h + (size_t)l * 3 * HID * HID, attn_in_b + l * 3 * HID,
        nullptr, qkv, nullptr, SEQ, 3 * HID, HID, 0);
    // attention -> attn_oh (f16)
    attn_kernel<<<dim3(SEQ, NHEAD), 256, 0, stream>>>(qkv, lengths, attn_oh);
    // x += attn_oh @ Wout^T + b
    gemm_wmma<<<dim3(HID / 64, SEQ / 32), 256, GEMM_LDS, stream>>>(
        attn_oh, Wout_h + (size_t)l * HID * HID, attn_out_b + l * HID,
        x, x, nullptr, SEQ, HID, HID, 0);
    // ln2 -> y_h
    ln_kernel<<<SEQ, HID, 0, stream>>>(x, ln2_s + l * HID, ln2_b + l * HID, y_h);
    // hid = relu(y @ W1^T + b1) -> f16
    gemm_wmma<<<dim3(DFFI / 64, SEQ / 32), 256, GEMM_LDS, stream>>>(
        y_h, W1_h + (size_t)l * DFFI * HID, ffn_b1 + l * DFFI,
        nullptr, nullptr, hid_h, SEQ, DFFI, HID, 1);
    // x += hid @ W2^T + b2
    gemm_wmma<<<dim3(HID / 64, SEQ / 32), 256, GEMM_LDS, stream>>>(
        hid_h, W2_h + (size_t)l * HID * DFFI, ffn_b2 + l * HID,
        x, x, nullptr, SEQ, HID, DFFI, 0);
  }

  // enc_out (f16 copy) and pointer projections
  cvt(x, enc_h, SEQ * HID);
  gemm_wmma<<<dim3(HID / 64, SEQ / 32), 256, GEMM_LDS, stream>>>(
      enc_h, gWref_h, nullptr, nullptr, ref_g, nullptr, SEQ, HID, HID, 0);
  gemm_wmma<<<dim3(HID / 64, SEQ / 32), 256, GEMM_LDS, stream>>>(
      enc_h, pWref_h, nullptr, nullptr, ref_p, nullptr, SEQ, HID, HID, 0);

  init_bar_kernel<<<1, 1, 0, stream>>>(bar);
  decode_kernel<<<NB, 256, 0, stream>>>(
      emb, x, ref_g, ref_p,
      lstm_Wih, lstm_Whh, lstm_bih, lstm_bhh,
      g_Wq, g_v, p_Wq, p_v, dec0, lengths,
      sgbuf, spbuf, bar, tour, logp);
}
